// VQQuantizer_44006234915601
// MI455X (gfx1250) — compile-verified, matrix operand order checked by reading, not run on hardware
//
#include <hip/hip_runtime.h>

// ---------------------------------------------------------------------------
// VQ quantizer, fused for MI455X (gfx1250, wave32, WMMA + async/TDM).
//   N=32768 tokens, D=512, K=2048 codes.
//   GEMM1: sim = hf(16xD) x cb^T  -> logits block [16 x 2048] in LDS
//   softmax/argmax in LDS, q emitted via TDM store (tensor_store_from_lds)
//   GEMM2: c_tilde = q(16x2048) x cb(2048xD) via transposed f16 codebook
// Both GEMMs use v_wmma_f32_16x16x32_f16; two independent accumulator chains
// per wave in GEMM1 (+reuse_a hint on the paired WMMA).
// ---------------------------------------------------------------------------

typedef __attribute__((ext_vector_type(16))) _Float16 v16h;
typedef __attribute__((ext_vector_type(8)))  float    v8f;
typedef __attribute__((ext_vector_type(4)))  unsigned v4u;
typedef __attribute__((ext_vector_type(8)))  int      v8i;
typedef __attribute__((ext_vector_type(4)))  int      v4i;

#define N_TOK 32768
#define DIM   512
#define KCB   2048
#define BM    16
#define NWAVE 8
#define TPB   256

// ---- gfx1250 feature probes (graceful fallback if a builtin is absent) ----
#if defined(__has_builtin)
#  if __has_builtin(__builtin_amdgcn_tensor_load_to_lds)
#    define HAS_TDM_LD 1
#  endif
#  if __has_builtin(__builtin_amdgcn_tensor_store_from_lds)
#    define HAS_TDM_ST 1
#  endif
#  if __has_builtin(__builtin_amdgcn_global_load_async_to_lds_b128)
#    define HAS_ASYNC 1
#  endif
#endif
#ifndef HAS_TDM_LD
#  define HAS_TDM_LD 0
#endif
#ifndef HAS_TDM_ST
#  define HAS_TDM_ST 0
#endif
#ifndef HAS_ASYNC
#  define HAS_ASYNC 0
#endif

// Typed pointer casts for the async builtin:
//   arg0: generic int4*  (global address), arg1: addrspace(3) int4* (LDS offset)
#define ASG4(p) ((v4i*)(unsigned long long)(p))
#define ASL4(p) ((__attribute__((address_space(3))) v4i*)(unsigned)(unsigned long long)(p))

#if HAS_ASYNC
#  if __has_builtin(__builtin_amdgcn_s_wait_asynccnt)
#    define WAIT_ASYNC(n) __builtin_amdgcn_s_wait_asynccnt(n)
#  else
#    define WAIT_ASYNC(n) asm volatile("s_wait_asynccnt %0" :: "i"(n) : "memory")
#  endif
#else
#  define WAIT_ASYNC(n)
#endif

union HV16 {
    v16h     v;
    _Float16 h[16];
    uint4    u[2];
};

#if HAS_TDM_LD || HAS_TDM_ST
// Build a 2D-tile D# (D# group0 + group1) for data_size=8B rows.
__device__ __forceinline__ void tdm_desc(unsigned lds, unsigned long long ga,
                                         unsigned d0, unsigned rows, unsigned stride,
                                         v4u* g0o, v8i* g1o)
{
    v4u g0;
    g0[0] = 1u;                                         // count=1, gather off
    g0[1] = lds;                                        // lds_addr
    g0[2] = (unsigned)ga;                               // global_addr[31:0]
    g0[3] = (unsigned)((ga >> 32) & 0x1FFFFFFull)       // global_addr[56:32]
            | (2u << 30);                               // type = image
    v8i g1;
    g1[0] = (int)(3u << 16);                            // data_size = 8 bytes
    g1[1] = (int)((d0 & 0xFFFFu) << 16);                // tensor_dim0[15:0]
    g1[2] = (int)(((d0 >> 16) & 0xFFFFu) |              // tensor_dim0[31:16]
                  ((rows & 0xFFFFu) << 16));            // tensor_dim1[15:0]
    g1[3] = (int)((d0 & 0xFFFFu) << 16);                // tile_dim0
    g1[4] = (int)(rows & 0xFFFFu);                      // tile_dim1 (tile_dim2=0)
    g1[5] = (int)stride;                                // tensor_dim0_stride[31:0]
    g1[6] = 0;
    g1[7] = 0;
    *g0o = g0;
    *g1o = g1;
}
#endif

#if HAS_TDM_LD
__device__ __forceinline__ void tdm_load_call(v4u g0, v8i g1)
{
    v4i z4 = {};
#if __clang_major__ >= 23
    v8i z8 = {};
    __builtin_amdgcn_tensor_load_to_lds(g0, g1, z4, z4, z8, 0);
#else
    __builtin_amdgcn_tensor_load_to_lds(g0, g1, z4, z4, 0);
#endif
}
#endif

#if HAS_TDM_ST
__device__ __forceinline__ void tdm_store_call(v4u g0, v8i g1)
{
    v4i z4 = {};
#if __clang_major__ >= 23
    v8i z8 = {};
    __builtin_amdgcn_tensor_store_from_lds(g0, g1, z4, z4, z8, 0);
#else
    __builtin_amdgcn_tensor_store_from_lds(g0, g1, z4, z4, 0);
#endif
}
#endif

#if HAS_ASYNC
// Async-stage one 32-k chunk of two 16-code B panels into LDS (4 x b128/lane).
__device__ __forceinline__ void stage2_async(const _Float16* __restrict__ cb16,
                                             int c0a, int c0b, int chunk,
                                             _Float16* b0, _Float16* b1, int lane)
{
    const int code = lane & 15;
    const int s0   = (lane >> 4) * 2;
    const char* ga = (const char*)(cb16 + (size_t)(c0a + code) * DIM + chunk * 32);
    const char* gb = (const char*)(cb16 + (size_t)(c0b + code) * DIM + chunk * 32);
    char* la = (char*)(b0 + code * 32);
    char* lb = (char*)(b1 + code * 32);
    __builtin_amdgcn_global_load_async_to_lds_b128(ASG4(ga + s0 * 16),      ASL4(la + s0 * 16),      0, 0);
    __builtin_amdgcn_global_load_async_to_lds_b128(ASG4(ga + s0 * 16 + 16), ASL4(la + s0 * 16 + 16), 0, 0);
    __builtin_amdgcn_global_load_async_to_lds_b128(ASG4(gb + s0 * 16),      ASL4(lb + s0 * 16),      0, 0);
    __builtin_amdgcn_global_load_async_to_lds_b128(ASG4(gb + s0 * 16 + 16), ASL4(lb + s0 * 16 + 16), 0, 0);
}
#endif

// ---------------------------------------------------------------------------
// Row L2-normalize h (one wave per row of 512 floats) -> f16 copy.
// ---------------------------------------------------------------------------
__global__ void vq_norm_h_kernel(const float* __restrict__ h,
                                 _Float16* __restrict__ h16)
{
    const int wave = threadIdx.x >> 5;
    const int lane = threadIdx.x & 31;
    const int row  = blockIdx.x * (TPB / 32) + wave;

    const float4* src = (const float4*)(h + (size_t)row * DIM);
    float4 vals[4];
    float ss = 0.f;
#pragma unroll
    for (int i = 0; i < 4; ++i) {
        vals[i] = src[lane * 4 + i];
        ss += vals[i].x * vals[i].x + vals[i].y * vals[i].y +
              vals[i].z * vals[i].z + vals[i].w * vals[i].w;
    }
#pragma unroll
    for (int off = 16; off > 0; off >>= 1) ss += __shfl_xor(ss, off, 32);
    const float inv = 1.0f / fmaxf(sqrtf(ss), 1e-6f);

    HV16 hv;
#pragma unroll
    for (int i = 0; i < 4; ++i) {
        hv.h[i * 4 + 0] = (_Float16)(vals[i].x * inv);
        hv.h[i * 4 + 1] = (_Float16)(vals[i].y * inv);
        hv.h[i * 4 + 2] = (_Float16)(vals[i].z * inv);
        hv.h[i * 4 + 3] = (_Float16)(vals[i].w * inv);
    }
    uint4* dst = (uint4*)(h16 + (size_t)row * DIM);
    dst[lane * 2 + 0] = hv.u[0];
    dst[lane * 2 + 1] = hv.u[1];
}

// ---------------------------------------------------------------------------
// Normalize codebook: f32 normalized copy, f16 copy (KxD), f16 transposed (DxK).
// ---------------------------------------------------------------------------
__global__ void vq_norm_cb_kernel(const float* __restrict__ cb,
                                  float* __restrict__ cbn,
                                  _Float16* __restrict__ cb16,
                                  _Float16* __restrict__ cbT16)
{
    const int wave = threadIdx.x >> 5;
    const int lane = threadIdx.x & 31;
    const int row  = blockIdx.x * (TPB / 32) + wave;

    const float4* src = (const float4*)(cb + (size_t)row * DIM);
    float4 vals[4];
    float ss = 0.f;
#pragma unroll
    for (int i = 0; i < 4; ++i) {
        vals[i] = src[lane * 4 + i];
        ss += vals[i].x * vals[i].x + vals[i].y * vals[i].y +
              vals[i].z * vals[i].z + vals[i].w * vals[i].w;
    }
#pragma unroll
    for (int off = 16; off > 0; off >>= 1) ss += __shfl_xor(ss, off, 32);
    const float inv = 1.0f / fmaxf(sqrtf(ss), 1e-6f);

    HV16 hv;
    float4* dstn = (float4*)(cbn + (size_t)row * DIM);
#pragma unroll
    for (int i = 0; i < 4; ++i) {
        float4 s;
        s.x = vals[i].x * inv; s.y = vals[i].y * inv;
        s.z = vals[i].z * inv; s.w = vals[i].w * inv;
        dstn[lane * 4 + i] = s;
        hv.h[i * 4 + 0] = (_Float16)s.x;
        hv.h[i * 4 + 1] = (_Float16)s.y;
        hv.h[i * 4 + 2] = (_Float16)s.z;
        hv.h[i * 4 + 3] = (_Float16)s.w;
    }
    uint4* dst16 = (uint4*)(cb16 + (size_t)row * DIM);
    dst16[lane * 2 + 0] = hv.u[0];
    dst16[lane * 2 + 1] = hv.u[1];
#pragma unroll
    for (int j = 0; j < 16; ++j)
        cbT16[(size_t)(lane * 16 + j) * KCB + row] = hv.h[j];
}

// ---------------------------------------------------------------------------
// Fused block kernel: one block = 16 token rows.
// ---------------------------------------------------------------------------
__global__ void __launch_bounds__(TPB)
vq_fused_kernel(const _Float16* __restrict__ h16,
                const _Float16* __restrict__ cb16,
                const _Float16* __restrict__ cbT16,
                const float*    __restrict__ cbn,
                const float*    __restrict__ u,
                float* __restrict__ out_q,
                float* __restrict__ out_ct,
                float* __restrict__ out_ch,
                float* __restrict__ out_cq,
                float* __restrict__ out_loss,
                float* __restrict__ out_idx)
{
    // Phase-disjoint LDS overlay: hf stage (ph0) / B staging (ph1) / idx+red (ph2+)
    __shared__ __align__(16) union {
        _Float16 A[BM * DIM];                      // 16 KB
        _Float16 B[NWAVE][2][2][BM * 32];          // 32 KB (parity x 2 tiles)
        struct { int idx[BM]; float red[NWAVE]; } s;
    } sU;
    __shared__ __align__(16) float sL[BM][KCB];    // 128 KB logits -> q

    const int tid  = threadIdx.x;
    const int wave = tid >> 5;
    const int lane = tid & 31;
    const int r0   = blockIdx.x * BM;

    // ---- phase 0: stage normalized token rows into LDS ----
#if HAS_TDM_LD
    if (wave == 0) {
        v4u g0; v8i g1;
        tdm_desc((unsigned)(unsigned long long)(uintptr_t)sU.A,
                 (unsigned long long)(uintptr_t)(h16 + (size_t)r0 * DIM),
                 128u /* 1024B rows in 8B units */, BM, 128u, &g0, &g1);
        tdm_load_call(g0, g1);
        __builtin_amdgcn_s_wait_tensorcnt(0);
    }
#else
    {
        const uint4* g = (const uint4*)(h16 + (size_t)r0 * DIM);
        uint4* sdst = (uint4*)sU.A;
#pragma unroll
        for (int i = 0; i < 4; ++i) sdst[tid * 4 + i] = g[tid * 4 + i];
    }
#endif
    __syncthreads();

    // ---- A fragments (16x32 f16 per k-chunk) resident in VGPRs ----
    const int m  = lane & 15;
    const int kb = (lane >> 4) * 8;
    v16h aFrag[16];
#pragma unroll
    for (int c = 0; c < 16; ++c) {
        HV16 hv;
        const uint4* sa = (const uint4*)(sU.A + m * DIM + c * 32 + kb);
        hv.u[0] = sa[0];
        hv.u[1] = sa[2];
        aFrag[c] = hv.v;
    }
    __syncthreads();   // sU.A dead; sU.B staging may begin

    // ======== phase 1: logits = 2*sim - 2 + gumbel, 2 tiles/pass ========
    for (int tp = 0; tp < 8; ++tp) {
        const int c0a = wave * 256 + tp * 32;
        const int c0b = c0a + 16;
        v8f accA = {}, accB = {};
#if HAS_ASYNC
        stage2_async(cb16, c0a, c0b, 0, sU.B[wave][0][0], sU.B[wave][0][1], lane);
#endif
#pragma unroll
        for (int c = 0; c < 16; ++c) {
#if HAS_ASYNC
            if (c < 15) {
                stage2_async(cb16, c0a, c0b, c + 1,
                             sU.B[wave][(c + 1) & 1][0], sU.B[wave][(c + 1) & 1][1], lane);
                WAIT_ASYNC(4);   // wait for current buffer; next stays in flight
            } else {
                WAIT_ASYNC(0);
            }
#else
            {
                const int code = lane & 15;
                const int s0   = (lane >> 4) * 2;
                const uint4* g0p = (const uint4*)(cb16 + (size_t)(c0a + code) * DIM + c * 32);
                const uint4* g1p = (const uint4*)(cb16 + (size_t)(c0b + code) * DIM + c * 32);
                uint4* l0 = (uint4*)(sU.B[wave][c & 1][0] + code * 32);
                uint4* l1 = (uint4*)(sU.B[wave][c & 1][1] + code * 32);
                l0[s0]     = g0p[s0];
                l0[s0 + 1] = g0p[s0 + 1];
                l1[s0]     = g1p[s0];
                l1[s0 + 1] = g1p[s0 + 1];
            }
            if (c < 15)
                __builtin_prefetch(cb16 + (size_t)(c0a + (lane & 15)) * DIM + (c + 1) * 32, 0, 1);
#endif
            const int buf = c & 1;
            HV16 bva, bvb;
            const uint4* sb0 = (const uint4*)(sU.B[wave][buf][0] + (lane & 15) * 32 + kb);
            const uint4* sb1 = (const uint4*)(sU.B[wave][buf][1] + (lane & 15) * 32 + kb);
            bva.u[0] = sb0[0]; bva.u[1] = sb0[2];
            bvb.u[0] = sb1[0]; bvb.u[1] = sb1[2];
            // two independent accumulator chains; reuse_a on the paired WMMA
            accA = __builtin_amdgcn_wmma_f32_16x16x32_f16(
                false, aFrag[c], false, bva.v, (short)0, accA, false, false);
            accB = __builtin_amdgcn_wmma_f32_16x16x32_f16(
                false, aFrag[c], false, bvb.v, (short)0, accB, true, false);
        }
#pragma unroll
        for (int j = 0; j < 8; ++j) {
            const int mm = j + ((lane >> 4) << 3);
            const int nn = lane & 15;
            const float ua = u[(size_t)(r0 + mm) * KCB + (c0a + nn)];
            const float ub = u[(size_t)(r0 + mm) * KCB + (c0b + nn)];
            sL[mm][c0a + nn] = 2.0f * accA[j] - 2.0f - __logf(-__logf(ua + 1e-10f) + 1e-10f);
            sL[mm][c0b + nn] = 2.0f * accB[j] - 2.0f - __logf(-__logf(ub + 1e-10f) + 1e-10f);
        }
    }
    __syncthreads();

    // ======== phase 2: softmax + argmax over K ========
    {
        const int row = tid >> 4;
        const int sub = tid & 15;       // 16 threads/row, 128 cols each
        float* Lrow = sL[row];
        float mx = -3.4e38f;
        int   ix = 0;
        for (int c2 = sub * 128; c2 < sub * 128 + 128; ++c2) {
            const float v = Lrow[c2];
            if (v > mx) { mx = v; ix = c2; }
        }
#pragma unroll
        for (int off = 8; off > 0; off >>= 1) {
            const float om = __shfl_xor(mx, off, 32);
            const int   oi = __shfl_xor(ix, off, 32);
            if (om > mx || (om == mx && oi < ix)) { mx = om; ix = oi; }
        }
        float sum = 0.f;
        for (int c2 = sub * 128; c2 < sub * 128 + 128; ++c2) {
            const float e = __expf(Lrow[c2] - mx);
            Lrow[c2] = e;
            sum += e;
        }
#pragma unroll
        for (int off = 8; off > 0; off >>= 1) sum += __shfl_xor(sum, off, 32);
        const float inv = 1.0f / sum;
#if HAS_TDM_ST
        for (int c2 = sub * 128; c2 < sub * 128 + 128; c2 += 4) {
            float4 e4 = *(float4*)&Lrow[c2];
            e4.x *= inv; e4.y *= inv; e4.z *= inv; e4.w *= inv;
            *(float4*)&Lrow[c2] = e4;        // q stays in LDS; TDM emits it below
        }
#else
        float* qg = out_q + (size_t)(r0 + row) * KCB;
        for (int c2 = sub * 128; c2 < sub * 128 + 128; c2 += 4) {
            float4 e4 = *(float4*)&Lrow[c2];
            e4.x *= inv; e4.y *= inv; e4.z *= inv; e4.w *= inv;
            *(float4*)&Lrow[c2] = e4;
            *(float4*)(qg + c2) = e4;
        }
#endif
        if (sub == 0) {
            sU.s.idx[row] = ix;
            out_idx[r0 + row] = (float)ix;
        }
    }
    __syncthreads();

#if HAS_TDM_ST
    // ---- emit the whole 16x2048 q block (128 KB contiguous) via the TDM ----
    if (wave == 0) {
        v4u g0; v8i g1;
        tdm_desc((unsigned)(unsigned long long)(uintptr_t)&sL[0][0],
                 (unsigned long long)(uintptr_t)(out_q + (size_t)r0 * KCB),
                 1024u /* 8192B rows in 8B units */, BM, 1024u, &g0, &g1);
        tdm_store_call(g0, g1);   // s_endpgm's implicit wait-idle covers TENSORcnt
    }
#endif

    // ---- c_hard / c_quantized = cbn[argmax] ----
    {
        const int row = tid >> 4;
        const int sub = tid & 15;
        const int idx = sU.s.idx[row];
        const float4* src = (const float4*)(cbn + (size_t)idx * DIM);
        float4* dh = (float4*)(out_ch + (size_t)(r0 + row) * DIM);
        float4* dq = (float4*)(out_cq + (size_t)(r0 + row) * DIM);
#pragma unroll
        for (int i = 0; i < 8; ++i) {
            const float4 v = src[sub * 8 + i];
            dh[sub * 8 + i] = v;
            dq[sub * 8 + i] = v;
        }
    }

    // ======== phase 3: c_tilde = q @ cbn (WMMA, via cbT16) ========
    float lossAcc = 0.f;
    {
        v8f zero = {};
        v8f acc2[4] = { zero, zero, zero, zero };   // this wave owns 4 d-tiles
        const int nn = lane & 15;
        for (int cc = 0; cc < 64; ++cc) {           // 2048 codes / 32
            HV16 qa;                                // q fragment: LDS f32 -> f16
            {
                const float* qp = &sL[m][cc * 32 + kb];
                float fr[16];
                *(float4*)&fr[0]  = *(const float4*)(qp + 0);
                *(float4*)&fr[4]  = *(const float4*)(qp + 4);
                *(float4*)&fr[8]  = *(const float4*)(qp + 16);
                *(float4*)&fr[12] = *(const float4*)(qp + 20);
#pragma unroll
                for (int j = 0; j < 16; ++j) qa.h[j] = (_Float16)fr[j];
            }
            // manual expansion so reuse_a is a frontend constant
#define G2_STEP(T, RA)                                                          \
            {                                                                   \
                const int d0 = (wave * 4 + (T)) * 16;                           \
                HV16 bv;                                                        \
                const uint4* g = (const uint4*)(cbT16 +                         \
                    (size_t)(d0 + nn) * KCB + cc * 32 + kb);                    \
                bv.u[0] = g[0];                                                 \
                bv.u[1] = g[2];                                                 \
                acc2[T] = __builtin_amdgcn_wmma_f32_16x16x32_f16(               \
                    false, qa.v, false, bv.v, (short)0, acc2[T], RA, false);    \
            }
            G2_STEP(0, false)
            G2_STEP(1, true)
            G2_STEP(2, true)
            G2_STEP(3, true)
#undef G2_STEP
        }
#pragma unroll
        for (int t = 0; t < 4; ++t) {
            const int d0 = (wave * 4 + t) * 16;
#pragma unroll
            for (int j = 0; j < 8; ++j) {
                const int mm = j + ((lane >> 4) << 3);
                const int dd = d0 + nn;
                const float ct = acc2[t][j];
                out_ct[(size_t)(r0 + mm) * DIM + dd] = ct;
                const float hf = (float)h16[(size_t)(r0 + mm) * DIM + dd];
                const float d  = ct - hf;
                lossAcc += d * d;
            }
        }
    }

    // ---- loss = (1+beta)*mean((c_tilde-hf)^2): shuffle + LDS + one atomic ----
#pragma unroll
    for (int off = 16; off > 0; off >>= 1) lossAcc += __shfl_xor(lossAcc, off, 32);
    if (lane == 0) sU.s.red[wave] = lossAcc;
    __syncthreads();
    if (tid == 0) {
        float tot = 0.f;
#pragma unroll
        for (int wv = 0; wv < NWAVE; ++wv) tot += sU.s.red[wv];
        atomicAdd(out_loss, tot * (1.25f / ((float)N_TOK * (float)DIM)));
    }
}

// ---------------------------------------------------------------------------
extern "C" void kernel_launch(void* const* d_in, const int* in_sizes, int n_in,
                              void* d_out, int out_size, void* d_ws, size_t ws_size,
                              hipStream_t stream)
{
    (void)in_sizes; (void)n_in; (void)out_size; (void)ws_size;

    const float* h  = (const float*)d_in[0];   // [16,2048,512]
    const float* cb = (const float*)d_in[1];   // [2048,512]
    const float* u  = (const float*)d_in[2];   // [32768,2048]

    char* ws = (char*)d_ws;
    _Float16* h16   = (_Float16*)(ws);                                 // 32 MB
    _Float16* cb16  = (_Float16*)(ws + (size_t)32 * 1024 * 1024);      //  2 MB
    _Float16* cbT16 = (_Float16*)(ws + (size_t)34 * 1024 * 1024);      //  2 MB
    float*    cbn   = (float*)   (ws + (size_t)36 * 1024 * 1024);      //  4 MB

    float* out      = (float*)d_out;
    float* out_q    = out;                                  // N*K
    float* out_ct   = out_q  + (size_t)N_TOK * KCB;         // N*D
    float* out_ch   = out_ct + (size_t)N_TOK * DIM;         // N*D
    float* out_cq   = out_ch + (size_t)N_TOK * DIM;         // N*D
    float* out_loss = out_cq + (size_t)N_TOK * DIM;         // 1
    float* out_idx  = out_loss + 1;                         // N

    (void)hipMemsetAsync(out_loss, 0, sizeof(float), stream);

    vq_norm_h_kernel <<<N_TOK / (TPB / 32), TPB, 0, stream>>>(h, h16);
    vq_norm_cb_kernel<<<KCB   / (TPB / 32), TPB, 0, stream>>>(cb, cbn, cb16, cbT16);
    vq_fused_kernel  <<<N_TOK / BM,         TPB, 0, stream>>>(
        h16, cb16, cbT16, cbn, u,
        out_q, out_ct, out_ch, out_cq, out_loss, out_idx);
}
